// TemporalDeformable_53068615910054
// MI455X (gfx1250) — matrix operand order repaired and checked by reading, not compiled, not run on hardware
//
#include <hip/hip_runtime.h>

// ---------------------------------------------------------------------------
// Problem constants (from reference)
// ---------------------------------------------------------------------------
#define BQ    2
#define DD    5
#define HH    32
#define WW    32
#define CC    256
#define NH    8
#define NL    5
#define NP    16
#define HD    32
#define MLPH  1024
#define HWSZ  1024          // H*W
#define LQ    5120          // D*H*W
#define NT    10240         // B*LQ  (tokens)

typedef __bf16 bf16;
typedef bf16  v16bf __attribute__((ext_vector_type(16)));
typedef bf16  v8bf  __attribute__((ext_vector_type(8)));
typedef float v8f   __attribute__((ext_vector_type(8)));
typedef int   v4i   __attribute__((ext_vector_type(4)));

// gfx1250 async global->LDS path (ASYNCcnt-tracked), if the toolchain has it.
#if __has_builtin(__builtin_amdgcn_global_load_async_to_lds_b128) && \
    __has_builtin(__builtin_amdgcn_s_wait_asynccnt)
#define HAVE_ASYNC_LDS 1
#else
#define HAVE_ASYNC_LDS 0
#endif

// ---------------------------------------------------------------------------
// Repack f32 weight (K x N, row-major) into WMMA bf16 B fragments, this file's contiguous-k order (same as A):
//   Bp[((nTile*ktiles + kTile)*32 + lane)*16 + e] =
//       W[(kTile*32 + 16*(lane>>4) + e) * N + nTile*16 + (lane&15)]
// so each lane fetches its whole 16-element fragment with two b128 loads.
// ---------------------------------------------------------------------------
__global__ __launch_bounds__(256)
void repack_w(const float* __restrict__ W, bf16* __restrict__ Wp, int K, int N) {
    int wid  = (blockIdx.x * blockDim.x + threadIdx.x) >> 5;
    int lane = threadIdx.x & 31;
    int ktiles = K >> 5;
    int total  = ktiles * (N >> 4);
    if (wid >= total) return;
    int nt = wid / ktiles;
    int kt = wid - nt * ktiles;
    int g  = lane >> 4;
    int n  = lane & 15;
    bf16* dst = Wp + ((size_t)wid * 32 + lane) * 16;
    const float* src = W + (size_t)(kt * 32 + 16 * g) * N + nt * 16 + n;
    #pragma unroll
    for (int e = 0; e < 16; ++e)
        dst[e] = (bf16)src[(size_t)e * N];
}

// ---------------------------------------------------------------------------
// LayerNorm over C=256 -> bf16. One block (256 thr) per token.
// wave32 shuffle reduce + LDS cross-wave combine.
// ---------------------------------------------------------------------------
__global__ __launch_bounds__(256)
void ln256_to_bf16(const float* __restrict__ x, const float* __restrict__ gam,
                   const float* __restrict__ bet, bf16* __restrict__ out) {
    int row = blockIdx.x, t = threadIdx.x;
    float v = x[(size_t)row * CC + t];
    float s = v, s2 = v * v;
    #pragma unroll
    for (int o = 16; o; o >>= 1) {
        s  += __shfl_xor(s,  o, 32);
        s2 += __shfl_xor(s2, o, 32);
    }
    __shared__ float r1[8], r2[8];
    if ((t & 31) == 0) { r1[t >> 5] = s; r2[t >> 5] = s2; }
    __syncthreads();
    float ts = 0.f, ts2 = 0.f;
    #pragma unroll
    for (int i = 0; i < 8; ++i) { ts += r1[i]; ts2 += r2[i]; }
    float mean = ts * (1.f / CC);
    float var  = ts2 * (1.f / CC) - mean * mean;
    float rs   = rsqrtf(var + 1e-5f);
    out[(size_t)row * CC + t] = (bf16)((v - mean) * rs * gam[t] + bet[t]);
}

// ---------------------------------------------------------------------------
// WMMA bf16 GEMM: C(M x N) = A(M x K, row-major bf16) * Bp(packed) + bias
//
// Block = 256 thr (8 waves) -> 128 rows x 64 cols.  Each wave owns a 16-row
// slice and 4 accumulators (16x64): one A fragment feeds 4 back-to-back
// v_wmma issues.  The 4 B column-tile strips for the current K-step (4 KB)
// are staged into double-buffered LDS once per block -- via
// GLOBAL_LOAD_ASYNC_TO_LDS_B128 (ASYNCcnt) when available, else plain
// b128 load + ds_store -- then every wave reads fragments with ds_load_b128.
//
// Epilogue modes:
//   0: f32 store: out = acc + bias (+ res)      (res optional residual)
//   1: bf16 store: out = gelu_exact(acc + bias)
//   2: f32 accumulate: out += acc + bias
// ---------------------------------------------------------------------------
__global__ __launch_bounds__(256)
void gemm_wmma_bf16(const bf16* __restrict__ A, const bf16* __restrict__ Bp,
                    const float* __restrict__ bias, const float* __restrict__ res,
                    void* __restrict__ out, int M, int N, int K, int mode) {
    __shared__ __align__(32) bf16 bsh[2][4 * 512];   // 2 x 4KB: 4 tiles x 32 lanes x 16

    int tid  = threadIdx.x;
    int wave = tid >> 5;
    int lane = tid & 31;
    int rowTile = blockIdx.y * 8 + wave;
    int nt0     = blockIdx.x * 4;          // first of 4 column tiles
    int ktiles  = K >> 5;

    int g  = lane >> 4;    // half-select
    int mn = lane & 15;    // row (for A) / col (for B & D) within tile

    const bf16* aRow = A + (size_t)(rowTile * 16 + mn) * K;

    // staging role: thread copies one 16B beat; tile chunk = tid/64
    int cchunk = tid >> 6;                 // 0..3 (column tile)
    int celem  = tid & 63;                 // 0..63 (64 x 16B = 1KB strip)
    const char* bsrc = (const char*)Bp;

    auto stage = [&](int kt, int buf) {
        const char* src = bsrc + (((size_t)(nt0 + cchunk) * ktiles + kt) << 10)
                               + ((size_t)celem << 4);
        bf16* dst = &bsh[buf][cchunk * 512 + celem * 8];
#if HAVE_ASYNC_LDS
        __builtin_amdgcn_global_load_async_to_lds_b128((v4i*)src, (v4i*)dst, 0, 0);
#else
        *reinterpret_cast<float4*>(dst) = *reinterpret_cast<const float4*>(src);
#endif
    };

    v8f acc0 = {}, acc1 = {}, acc2 = {}, acc3 = {};

    stage(0, 0);
#if HAVE_ASYNC_LDS
    __builtin_amdgcn_s_wait_asynccnt(0);
#endif
    __syncthreads();

    for (int kt = 0; kt < ktiles; ++kt) {
        int cur = kt & 1;
        if (kt + 1 < ktiles) {
            stage(kt + 1, cur ^ 1);
            __builtin_prefetch(aRow + (kt + 1) * 32 + 8 * g, 0, 1);
        }

        // A fragment: elems 0..15 = K [16g .. 16g+15]
        const bf16* ap = aRow + kt * 32 + 16 * g;
        v8bf alo = *reinterpret_cast<const v8bf*>(ap);
        v8bf ahi = *reinterpret_cast<const v8bf*>(ap + 8);
        v16bf a = __builtin_shufflevector(alo, ahi,
                                          0, 1, 2, 3, 4, 5, 6, 7,
                                          8, 9, 10, 11, 12, 13, 14, 15);

        // B fragments from LDS (ds_load_b128 x2 each)
        const bf16* bb = &bsh[cur][lane * 16];
        v16bf b0 = *reinterpret_cast<const v16bf*>(bb);
        v16bf b1 = *reinterpret_cast<const v16bf*>(bb + 512);
        v16bf b2 = *reinterpret_cast<const v16bf*>(bb + 1024);
        v16bf b3 = *reinterpret_cast<const v16bf*>(bb + 1536);

        acc0 = __builtin_amdgcn_wmma_f32_16x16x32_bf16(
                   false, a, false, b0, (short)0, acc0, false, false);
        acc1 = __builtin_amdgcn_wmma_f32_16x16x32_bf16(
                   false, a, false, b1, (short)0, acc1, false, false);
        acc2 = __builtin_amdgcn_wmma_f32_16x16x32_bf16(
                   false, a, false, b2, (short)0, acc2, false, false);
        acc3 = __builtin_amdgcn_wmma_f32_16x16x32_bf16(
                   false, a, false, b3, (short)0, acc3, false, false);

#if HAVE_ASYNC_LDS
        __builtin_amdgcn_s_wait_asynccnt(0);
#endif
        __syncthreads();
    }

    if (rowTile * 16 >= M) return;   // (all our M are multiples of 128)

    v8f accs[4] = {acc0, acc1, acc2, acc3};
    #pragma unroll
    for (int ct = 0; ct < 4; ++ct) {
        int col = (nt0 + ct) * 16 + mn;
        float bs = bias ? bias[col] : 0.f;
        #pragma unroll
        for (int j = 0; j < 8; ++j) {
            int row = rowTile * 16 + 8 * g + j;   // D layout: VGPR j -> M = j + 8*g
            size_t idx = (size_t)row * N + col;
            float v = accs[ct][j] + bs;
            if (mode == 0) {
                float r = res ? res[idx] : 0.f;
                reinterpret_cast<float*>(out)[idx] = v + r;
            } else if (mode == 1) {
                float gl = 0.5f * v * (1.f + erff(v * 0.70710678118654752f));
                reinterpret_cast<bf16*>(out)[idx] = (bf16)gl;
            } else {
                reinterpret_cast<float*>(out)[idx] += v;
            }
        }
    }
}

// ---------------------------------------------------------------------------
// Softmax over 80 (NL*NP) per head, in-place on f32 logits (NT x 640).
// One wave per (token, head): lanes cover elems {l, l+32, l+64}.
// ---------------------------------------------------------------------------
__global__ __launch_bounds__(256)
void softmax80(float* __restrict__ a) {
    int tok  = blockIdx.x;
    int h    = threadIdx.x >> 5;
    int lane = threadIdx.x & 31;
    float* p = a + (size_t)tok * (NH * NL * NP) + h * (NL * NP);
    float v0 = p[lane], v1 = p[lane + 32];
    float v2 = (lane < 16) ? p[lane + 64] : -3.0e38f;
    float m = fmaxf(v0, fmaxf(v1, v2));
    #pragma unroll
    for (int o = 16; o; o >>= 1) m = fmaxf(m, __shfl_xor(m, o, 32));
    float e0 = __expf(v0 - m), e1 = __expf(v1 - m);
    float e2 = (lane < 16) ? __expf(v2 - m) : 0.f;
    float s = e0 + e1 + e2;
    #pragma unroll
    for (int o = 16; o; o >>= 1) s += __shfl_xor(s, o, 32);
    float inv = 1.f / s;
    p[lane]      = e0 * inv;
    p[lane + 32] = e1 * inv;
    if (lane < 16) p[lane + 64] = e2 * inv;
}

// ---------------------------------------------------------------------------
// MSDeform bilinear sampling. One wave per (token, head); lane = head-dim
// channel -> every tap is one fully-coalesced 128B gather from val (L2-resident).
// ---------------------------------------------------------------------------
__global__ __launch_bounds__(256)
void msdeform_sample(const float* __restrict__ ref, const float* __restrict__ off,
                     const float* __restrict__ aw, const float* __restrict__ val,
                     bf16* __restrict__ samp) {
    int tok  = blockIdx.x;
    int h    = threadIdx.x >> 5;
    int lane = threadIdx.x & 31;
    int b = tok / LQ;

    const float* refp = ref + (size_t)tok * NL * 2;
    const float* offp = off + (size_t)tok * (NH * NL * NP * 2) + h * (NL * NP * 2);
    const float* awp  = aw  + (size_t)tok * (NH * NL * NP)     + h * (NL * NP);

    float acc = 0.f;
    for (int l = 0; l < NL; ++l) {
        float rx = refp[l * 2 + 0];
        float ry = refp[l * 2 + 1];
        const float* vl = val + ((size_t)b * LQ + (size_t)l * HWSZ) * CC + h * HD + lane;
        #pragma unroll 4
        for (int p = 0; p < NP; ++p) {
            float ox = offp[(l * NP + p) * 2 + 0];
            float oy = offp[(l * NP + p) * 2 + 1];
            float w  = awp[l * NP + p];
            // loc = ref + off/[W,H]; pixel = loc*size - 0.5
            float px = rx * (float)WW + ox - 0.5f;
            float py = ry * (float)HH + oy - 0.5f;
            float fx0 = floorf(px), fy0 = floorf(py);
            float fx = px - fx0, fy = py - fy0;
            int x0 = (int)fx0, y0 = (int)fy0;
            auto tap = [&](int yi, int xi, float wt) {
                if (xi >= 0 && xi < WW && yi >= 0 && yi < HH)
                    acc += w * wt * vl[(size_t)(yi * WW + xi) * CC];
            };
            tap(y0,     x0,     (1.f - fy) * (1.f - fx));
            tap(y0,     x0 + 1, (1.f - fy) * fx);
            tap(y0 + 1, x0,     fy * (1.f - fx));
            tap(y0 + 1, x0 + 1, fy * fx);
        }
    }
    samp[(size_t)tok * CC + h * HD + lane] = (bf16)acc;
}

// ---------------------------------------------------------------------------
// Host-side orchestration
// ---------------------------------------------------------------------------
static inline size_t alignup(size_t v) { return (v + 255) & ~(size_t)255; }

extern "C" void kernel_launch(void* const* d_in, const int* in_sizes, int n_in,
                              void* d_out, int out_size, void* d_ws, size_t ws_size,
                              hipStream_t stream) {
    const float* x      = (const float*)d_in[0];
    const float* refpts = (const float*)d_in[1];
    // d_in[2] spatial_shapes, d_in[3] level_start_index: constant, hardcoded
    const float* n3g = (const float*)d_in[4];
    const float* n3b = (const float*)d_in[5];
    const float* n4g = (const float*)d_in[6];
    const float* n4b = (const float*)d_in[7];
    const float* w_off  = (const float*)d_in[8];
    const float* b_off  = (const float*)d_in[9];
    const float* w_attn = (const float*)d_in[10];
    const float* b_attn = (const float*)d_in[11];
    const float* w_val  = (const float*)d_in[12];
    const float* b_val  = (const float*)d_in[13];
    const float* w_out  = (const float*)d_in[14];
    const float* b_out  = (const float*)d_in[15];
    const float* w_fc1  = (const float*)d_in[16];
    const float* b_fc1  = (const float*)d_in[17];
    const float* w_fc2  = (const float*)d_in[18];
    const float* b_fc2  = (const float*)d_in[19];
    float* out = (float*)d_out;

    // ---- workspace carve-up ------------------------------------------------
    char* w = (char*)d_ws;
    size_t o = 0;
    bf16*  qbf    = (bf16*)(w + o); o += alignup((size_t)NT * CC * 2);
    bf16*  hbf    = (bf16*)(w + o); o += alignup((size_t)NT * CC * 2);
    bf16*  sampbf = (bf16*)(w + o); o += alignup((size_t)NT * CC * 2);
    bf16*  gbf    = (bf16*)(w + o); o += alignup((size_t)NT * MLPH * 2);
    float* valf   = (float*)(w + o); o += alignup((size_t)NT * CC * 4);
    float* offf   = (float*)(w + o); o += alignup((size_t)NT * (NH * NL * NP * 2) * 4);
    float* attnf  = (float*)(w + o); o += alignup((size_t)NT * (NH * NL * NP) * 4);
    bf16*  wvalp  = (bf16*)(w + o); o += alignup((size_t)CC * CC * 2);
    bf16*  woffp  = (bf16*)(w + o); o += alignup((size_t)CC * (NH * NL * NP * 2) * 2);
    bf16*  wattnp = (bf16*)(w + o); o += alignup((size_t)CC * (NH * NL * NP) * 2);
    bf16*  woutp  = (bf16*)(w + o); o += alignup((size_t)CC * CC * 2);
    bf16*  wfc1p  = (bf16*)(w + o); o += alignup((size_t)CC * MLPH * 2);
    bf16*  wfc2p  = (bf16*)(w + o); o += alignup((size_t)MLPH * CC * 2);
    (void)ws_size; (void)in_sizes; (void)n_in; (void)out_size;

    const int NOFF  = NH * NL * NP * 2;  // 1280
    const int NATTN = NH * NL * NP;      // 640

    // ---- 0. repack weights to WMMA B-fragment layout (bf16) ---------------
    auto rblocks = [](int K, int N) { return ((K >> 5) * (N >> 4) + 7) / 8; };
    repack_w<<<rblocks(CC, CC),    256, 0, stream>>>(w_val,  wvalp,  CC, CC);
    repack_w<<<rblocks(CC, NOFF),  256, 0, stream>>>(w_off,  woffp,  CC, NOFF);
    repack_w<<<rblocks(CC, NATTN), 256, 0, stream>>>(w_attn, wattnp, CC, NATTN);
    repack_w<<<rblocks(CC, CC),    256, 0, stream>>>(w_out,  woutp,  CC, CC);
    repack_w<<<rblocks(CC, MLPH),  256, 0, stream>>>(w_fc1,  wfc1p,  CC, MLPH);
    repack_w<<<rblocks(MLPH, CC),  256, 0, stream>>>(w_fc2,  wfc2p,  MLPH, CC);

    // ---- 1. norm3 -> q (bf16) ---------------------------------------------
    ln256_to_bf16<<<NT, 256, 0, stream>>>(x, n3g, n3b, qbf);

    // ---- 2. q-side projections (bf16 WMMA, f32 accum) ---------------------
    dim3 blk(256);
    gemm_wmma_bf16<<<dim3(CC / 64,    NT / 128), blk, 0, stream>>>(
        qbf, wvalp,  b_val,  nullptr, valf,  NT, CC,    CC, 0);
    gemm_wmma_bf16<<<dim3(NOFF / 64,  NT / 128), blk, 0, stream>>>(
        qbf, woffp,  b_off,  nullptr, offf,  NT, NOFF,  CC, 0);
    gemm_wmma_bf16<<<dim3(NATTN / 64, NT / 128), blk, 0, stream>>>(
        qbf, wattnp, b_attn, nullptr, attnf, NT, NATTN, CC, 0);

    // ---- 3. softmax over NL*NP per head -----------------------------------
    softmax80<<<NT, 256, 0, stream>>>(attnf);

    // ---- 4. deformable bilinear sampling ----------------------------------
    msdeform_sample<<<NT, 256, 0, stream>>>(refpts, offf, attnf, valf, sampbf);

    // ---- 5. output projection + attention residual: d_out = x + samp@Wout --
    gemm_wmma_bf16<<<dim3(CC / 64, NT / 128), blk, 0, stream>>>(
        sampbf, woutp, b_out, x, out, NT, CC, CC, 0);

    // ---- 6. norm4 -> h (bf16) ---------------------------------------------
    ln256_to_bf16<<<NT, 256, 0, stream>>>(out, n4g, n4b, hbf);

    // ---- 7. MLP: gelu(h@fc1) (bf16) then d_out += g@fc2 --------------------
    gemm_wmma_bf16<<<dim3(MLPH / 64, NT / 128), blk, 0, stream>>>(
        hbf, wfc1p, b_fc1, nullptr, gbf, NT, MLPH, CC, 1);
    gemm_wmma_bf16<<<dim3(CC / 64, NT / 128), blk, 0, stream>>>(
        gbf, wfc2p, b_fc2, nullptr, out, NT, CC, MLPH, 2);
}
